// SPCNNRenderer_82471962018591
// MI455X (gfx1250) — matrix operand order repaired and checked by reading, not compiled, not run on hardware
//
#include <hip/hip_runtime.h>
#include <stdint.h>
#include <math.h>

#define BDIM 4
#define HDIM 256
#define WDIM 256
#define NPTS (BDIM * 32768) /* 131072 active sites */

typedef __bf16 bf16;
typedef bf16  bf16x16 __attribute__((ext_vector_type(16)));
typedef float f32x8   __attribute__((ext_vector_type(8)));

enum { EPI_NONE = 0, EPI_RELU = 1, EPI_RELU_RES = 2, EPI_SIGMOID = 3 };

// ---------------------------------------------------------------- utilities

__global__ void fill_map_kernel(int* __restrict__ map, int n) {
  int i = blockIdx.x * blockDim.x + threadIdx.x;
  if (i < n) map[i] = -1;
}

__global__ void scatter_map_kernel(const int* __restrict__ sidx, int* __restrict__ map, int n) {
  int i = blockIdx.x * blockDim.x + threadIdx.x;
  if (i < n) {
    int b = sidx[3 * i], y = sidx[3 * i + 1], x = sidx[3 * i + 2];
    map[(b * HDIM + y) * WDIM + x] = i;
  }
}

__global__ void zero_f32_kernel(float* __restrict__ p, int n) {
  int i = blockIdx.x * blockDim.x + threadIdx.x;
  if (i < n) p[i] = 0.f;
}

// fp32 weight [K, Cout] row-major  ->  bf16 transposed [Cpad, K], zero-pad cols >= Cout.
__global__ void convert_wT_kernel(const float* __restrict__ src, bf16* __restrict__ dst,
                                  int K, int Cout, int Cpad) {
  int i = blockIdx.x * blockDim.x + threadIdx.x;
  if (i >= K * Cpad) return;
  int c = i % Cpad;
  int k = i / Cpad;
  float v = (c < Cout) ? src[k * Cout + c] : 0.f;
  dst[(size_t)c * K + k] = (bf16)v;
}

// --------------------------------------------------- fused gather-GEMM-WMMA
//
// Block computes 64 output rows (4 row tiles) x COUT columns.
// Wave w owns col tiles [w*TPW, (w+1)*TPW); it keeps B fragments in registers
// and sweeps all 4 row tiles, so each B load feeds 4 WMMAs.
//
// A fragment layout (16-bit, 16x32): lane<16 -> row=lane,    K = {k0..k0+7, k0+16..k0+23}
//                                    lane>=16 -> row=lane-16, K = {k0+8..k0+15, k0+24..k0+31}
// B fragment layout (16-bit, 32x16): lane<16 -> col=lane,    K = {k0..k0+7, k0+16..k0+23}
//                                    lane>=16 -> col=lane-16, K = {k0+8..k0+15, k0+24..k0+31}
//
// TAPS==1: A loaded per-lane straight from global (each lane owns one row) - no LDS.
// TAPS==9: rows are gathered via the site map and staged in LDS with
//          global_load_async_to_lds_b128 (ASYNCcnt), shared by all waves.

template <int CIN, int COUT, int TAPS, int EPI, bool A_F32>
__global__ __launch_bounds__(128)
void spconv_wmma_kernel(const void* __restrict__ in_v,
                        const bf16* __restrict__ wT,   // [COUTpad][TAPS*CIN]
                        const float* __restrict__ bias,
                        const bf16* __restrict__ res,  // residual [N, COUT] (EPI_RELU_RES)
                        const int* __restrict__ sidx,  // [N,3]
                        const int* __restrict__ map,   // [B*H*W]
                        bf16* __restrict__ out,        // [N, COUT]
                        float* __restrict__ outf) {    // dense (B,3,H,W) for EPI_SIGMOID
  constexpr int NT = COUT / 16;
  constexpr int WAVES = (NT < 4) ? NT : 4;
  constexpr int TPW = NT / WAVES;
  constexpr int Ktot = TAPS * CIN;
  constexpr int NTHREADS = WAVES * 32;
  constexpr int MT = 4;  // row tiles per block -> M = 64

  const int m0 = blockIdx.x * (MT * 16);
  const int tid = threadIdx.x;
  const int wav = tid >> 5;
  const int lane = tid & 31;
  const int half = lane >> 4;
  const int l16 = lane & 15;

  constexpr int LDSA = (TAPS > 1) ? (MT * 16 * CIN) : 16;
  __shared__ __align__(16) bf16 As[LDSA];
  __shared__ int grow[(TAPS > 1) ? MT * 16 : 1];

  const f32x8 zero8 = {0.f, 0.f, 0.f, 0.f, 0.f, 0.f, 0.f, 0.f};
  f32x8 acc[MT][TPW];
#pragma unroll
  for (int rt = 0; rt < MT; ++rt)
#pragma unroll
    for (int j = 0; j < TPW; ++j) acc[rt][j] = zero8;

  for (int tap = 0; tap < TAPS; ++tap) {
    if (TAPS > 1) {
      // ---- gather the 64 input rows for this tap
      if (tid < MT * 16) {
        int s = m0 + tid;
        int bb = sidx[3 * s], yy = sidx[3 * s + 1], xx = sidx[3 * s + 2];
        int ny = yy + (tap / 3) - 1;
        int nx = xx + (tap % 3) - 1;
        grow[tid] = (ny >= 0 && ny < HDIM && nx >= 0 && nx < WDIM)
                        ? map[(bb * HDIM + ny) * WDIM + nx]
                        : -1;
      }
      __syncthreads();

      // ---- async-stage A (64 x CIN bf16) into LDS; zero-fill missing rows
      constexpr int CPR = CIN / 8;  // 16B chunks per row
      for (int c = tid; c < MT * 16 * CPR; c += NTHREADS) {
        int r = c / CPR;
        int off = (c % CPR) * 8;
        int g = grow[r];
        bf16* dstp = &As[r * CIN + off];
        if (g >= 0) {
          const bf16* srcp = (const bf16*)in_v + (size_t)g * CIN + off;
          unsigned ldsoff = (unsigned)(uintptr_t)dstp;  // flat LDS addr keeps offset in [31:0]
          asm volatile("global_load_async_to_lds_b128 %0, %1, off"
                       :
                       : "v"(ldsoff), "v"(srcp)
                       : "memory");
        } else {
          uint4 z;
          z.x = z.y = z.z = z.w = 0u;
          *(uint4*)dstp = z;
        }
      }
      asm volatile("s_wait_asynccnt 0x0" ::: "memory");
      __syncthreads();
    }

    // ---- K loop: 32 K-values per WMMA
    for (int k0 = 0; k0 < CIN; k0 += 32) {
      bf16x16 a[MT];
#pragma unroll
      for (int rt = 0; rt < MT; ++rt) {
        if (TAPS > 1) {
          const bf16* p = &As[(rt * 16 + l16) * CIN + k0 + half * 8];
          ((uint4*)&a[rt])[0] = *(const uint4*)p;
          ((uint4*)&a[rt])[1] = *(const uint4*)(p + 16);
        } else if (A_F32) {
          const float* p = (const float*)in_v +
                           (size_t)(m0 + rt * 16 + l16) * CIN + k0 + half * 8;
#pragma unroll
          for (int e = 0; e < 8; ++e) a[rt][e] = (bf16)p[e];
#pragma unroll
          for (int e = 0; e < 8; ++e) a[rt][8 + e] = (bf16)p[16 + e];
        } else {
          const bf16* p = (const bf16*)in_v +
                          (size_t)(m0 + rt * 16 + l16) * CIN + k0 + half * 8;
          ((uint4*)&a[rt])[0] = *(const uint4*)p;
          ((uint4*)&a[rt])[1] = *(const uint4*)(p + 16);
        }
      }
#pragma unroll
      for (int j = 0; j < TPW; ++j) {
        int col = (wav * TPW + j) * 16 + l16;
        const bf16* bp = wT + (size_t)col * Ktot + tap * CIN + k0 + half * 8;
        bf16x16 b;
        ((uint4*)&b)[0] = ((const uint4*)bp)[0];
        ((uint4*)&b)[1] = ((const uint4*)(bp + 16))[0];
#pragma unroll
        for (int rt = 0; rt < MT; ++rt) {
          acc[rt][j] = __builtin_amdgcn_wmma_f32_16x16x32_bf16(
              false, a[rt], false, b, (short)0, acc[rt][j], false, false);
        }
      }
    }
    if (TAPS > 1) __syncthreads();
  }

  // ---- epilogue (C/D layout: lane<16 -> col=lane, rows 0..7; lane>=16 -> col=lane-16, rows 8..15)
#pragma unroll
  for (int rt = 0; rt < MT; ++rt) {
#pragma unroll
    for (int j = 0; j < TPW; ++j) {
      int col = (wav * TPW + j) * 16 + l16;
#pragma unroll
      for (int e = 0; e < 8; ++e) {
        int srow = m0 + rt * 16 + e + half * 8;
        float v = acc[rt][j][e];
        if (EPI == EPI_SIGMOID) {
          if (col < 3) {
            v += bias[col];
            v = 1.f / (1.f + __expf(-v));
            int bb = sidx[3 * srow], yy = sidx[3 * srow + 1], xx = sidx[3 * srow + 2];
            outf[(((size_t)bb * 3 + col) * HDIM + yy) * WDIM + xx] = v;
          }
        } else {
          if (EPI != EPI_NONE) v += bias[col];
          if (EPI == EPI_RELU_RES) v += (float)res[(size_t)srow * COUT + col];
          if (EPI != EPI_NONE) v = (v > 0.f) ? v : 0.f;
          out[(size_t)srow * COUT + col] = (bf16)v;
        }
      }
    }
  }
}

// ------------------------------------------------------------------- launch

extern "C" void kernel_launch(void* const* d_in, const int* in_sizes, int n_in,
                              void* d_out, int out_size, void* d_ws, size_t ws_size,
                              hipStream_t stream) {
  (void)in_sizes; (void)n_in; (void)ws_size;

  const float* x    = (const float*)d_in[0];
  const int*   sidx = (const int*)d_in[1];
  const float* w0   = (const float*)d_in[2];
  const float* b0   = (const float*)d_in[3];
  const float* w1a  = (const float*)d_in[4];
  const float* b1a  = (const float*)d_in[5];
  const float* w1b  = (const float*)d_in[6];
  const float* b1b  = (const float*)d_in[7];
  const float* w1d  = (const float*)d_in[8];
  const float* w2a  = (const float*)d_in[9];
  const float* b2a  = (const float*)d_in[10];
  const float* w2b  = (const float*)d_in[11];
  const float* b2b  = (const float*)d_in[12];
  const float* w2d  = (const float*)d_in[13];
  const float* w3a  = (const float*)d_in[14];
  const float* b3a  = (const float*)d_in[15];
  const float* w3b  = (const float*)d_in[16];
  const float* b3b  = (const float*)d_in[17];
  const float* w3d  = (const float*)d_in[18];
  const float* w4   = (const float*)d_in[19];
  const float* b4   = (const float*)d_in[20];

  char* ws = (char*)d_ws;
  const size_t MB = 1u << 20;

  int* map = (int*)ws;  // 1 MB (B*H*W*4)

  bf16* wp = (bf16*)(ws + 1 * MB);  // ~1.3 MB of bf16 transposed weights
  bf16* w0T  = wp; wp += (size_t)256 * 256;
  bf16* w1aT = wp; wp += (size_t)128 * 2304;
  bf16* w1bT = wp; wp += (size_t)128 * 1152;
  bf16* w1dT = wp; wp += (size_t)128 * 256;
  bf16* w2aT = wp; wp += (size_t)64 * 1152;
  bf16* w2bT = wp; wp += (size_t)64 * 576;
  bf16* w2dT = wp; wp += (size_t)64 * 128;
  bf16* w3aT = wp; wp += (size_t)32 * 64;
  bf16* w3bT = wp; wp += (size_t)32 * 32;
  bf16* w3dT = wp; wp += (size_t)32 * 64;
  bf16* w4T  = wp; wp += (size_t)16 * 32;

  // feature buffers (bf16), with liveness-based reuse; peak ws use = 132 MB
  bf16* f0  = (bf16*)(ws + 4 * MB);    // [N,256]  64 MB
  bf16* f1a = (bf16*)(ws + 68 * MB);   // [N,128]  32 MB
  bf16* r1  = (bf16*)(ws + 100 * MB);  // [N,128]  32 MB
  bf16* f1  = (bf16*)(ws + 4 * MB);    // [N,128]  (f0 dead)
  bf16* f2a = (bf16*)(ws + 68 * MB);   // [N,64]   (f1a dead)
  bf16* r2  = (bf16*)(ws + 84 * MB);   // [N,64]
  bf16* f2  = (bf16*)(ws + 4 * MB);    // [N,64]   (f1 dead)
  bf16* f3a = (bf16*)(ws + 68 * MB);   // [N,32]
  bf16* r3  = (bf16*)(ws + 76 * MB);   // [N,32]
  bf16* f3  = (bf16*)(ws + 4 * MB);    // [N,32]   (f2 dead)

  const int npix = BDIM * HDIM * WDIM;
  fill_map_kernel<<<(npix + 255) / 256, 256, 0, stream>>>(map, npix);
  scatter_map_kernel<<<(NPTS + 255) / 256, 256, 0, stream>>>(sidx, map, NPTS);

  auto cvt = [&](const float* s, bf16* d, int K, int Cout, int Cpad) {
    int tot = K * Cpad;
    convert_wT_kernel<<<(tot + 255) / 256, 256, 0, stream>>>(s, d, K, Cout, Cpad);
  };
  cvt(w0,  w0T,  256,  256, 256);
  cvt(w1a, w1aT, 2304, 128, 128);
  cvt(w1b, w1bT, 1152, 128, 128);
  cvt(w1d, w1dT, 256,  128, 128);
  cvt(w2a, w2aT, 1152, 64,  64);
  cvt(w2b, w2bT, 576,  64,  64);
  cvt(w2d, w2dT, 128,  64,  64);
  cvt(w3a, w3aT, 64,   32,  32);
  cvt(w3b, w3bT, 32,   32,  32);
  cvt(w3d, w3dT, 64,   32,  32);
  cvt(w4,  w4T,  32,   3,   16);

  zero_f32_kernel<<<(out_size + 255) / 256, 256, 0, stream>>>((float*)d_out, out_size);

  dim3 grid(NPTS / 64);

  // layer 0: f0 = relu(x @ w0 + b0)            (A read as fp32, converted on load)
  spconv_wmma_kernel<256, 256, 1, EPI_RELU, true>
      <<<grid, 128, 0, stream>>>(x, w0T, b0, nullptr, sidx, map, f0, nullptr);
  // block 1
  spconv_wmma_kernel<256, 128, 9, EPI_RELU, false>
      <<<grid, 128, 0, stream>>>(f0, w1aT, b1a, nullptr, sidx, map, f1a, nullptr);
  spconv_wmma_kernel<256, 128, 1, EPI_NONE, false>
      <<<grid, 128, 0, stream>>>(f0, w1dT, nullptr, nullptr, sidx, map, r1, nullptr);
  spconv_wmma_kernel<128, 128, 9, EPI_RELU_RES, false>
      <<<grid, 128, 0, stream>>>(f1a, w1bT, b1b, r1, sidx, map, f1, nullptr);
  // block 2
  spconv_wmma_kernel<128, 64, 9, EPI_RELU, false>
      <<<grid, 128, 0, stream>>>(f1, w2aT, b2a, nullptr, sidx, map, f2a, nullptr);
  spconv_wmma_kernel<128, 64, 1, EPI_NONE, false>
      <<<grid, 128, 0, stream>>>(f1, w2dT, nullptr, nullptr, sidx, map, r2, nullptr);
  spconv_wmma_kernel<64, 64, 9, EPI_RELU_RES, false>
      <<<grid, 128, 0, stream>>>(f2a, w2bT, b2b, r2, sidx, map, f2, nullptr);
  // block 3 (all 1x1)
  spconv_wmma_kernel<64, 32, 1, EPI_RELU, false>
      <<<grid, 64, 0, stream>>>(f2, w3aT, b3a, nullptr, sidx, map, f3a, nullptr);
  spconv_wmma_kernel<64, 32, 1, EPI_NONE, false>
      <<<grid, 64, 0, stream>>>(f2, w3dT, nullptr, nullptr, sidx, map, r3, nullptr);
  spconv_wmma_kernel<32, 32, 1, EPI_RELU_RES, false>
      <<<grid, 64, 0, stream>>>(f3a, w3bT, b3b, r3, sidx, map, f3, nullptr);
  // final: sigmoid + scatter to dense (B,3,H,W)
  spconv_wmma_kernel<32, 16, 1, EPI_SIGMOID, false>
      <<<grid, 32, 0, stream>>>(f3, w4T, b4, nullptr, sidx, map, nullptr, (float*)d_out);
}
